// KeyJointACTGenerator_16922171147089
// MI455X (gfx1250) — compile-verified, matrix operand order checked by reading, not run on hardware
//
#include <hip/hip_runtime.h>
#include <cmath>

typedef _Float16 half8 __attribute__((ext_vector_type(8)));
typedef _Float16 v16h  __attribute__((ext_vector_type(16)));
typedef float    v8f   __attribute__((ext_vector_type(8)));
typedef unsigned int u32x4 __attribute__((ext_vector_type(4)));
typedef int          i32x4 __attribute__((ext_vector_type(4)));
typedef int          i32x8 __attribute__((ext_vector_type(8)));

#define DEV_INLINE __device__ __forceinline__

#if defined(__HIP_DEVICE_COMPILE__) && __has_builtin(__builtin_amdgcn_tensor_load_to_lds)
#define HAVE_TDM 1
#else
#define HAVE_TDM 0
#endif

// ---- model dims ----
constexpr int Bx    = 1024;
constexpr int ACTn  = 32;
constexpr int En    = 16;      // experts
constexpr int Tn    = 32;
constexpr int Dm    = 704;     // temporal input size
constexpr int NHn   = 8;
constexpr int HDn   = 88;
constexpr int TOK   = Bx * Tn; // 32768 tokens
constexpr int KC    = 64;      // K-chunk staged in LDS by the TDM

// =====================================================================
// Weight convert: f32 [K,N] (optionally E stacked) -> f16 transposed [N,Kp]
// zero-padded for k in [K,Kp) so every GEMM has K % 32 == 0.
// =====================================================================
__global__ void k_convT(const float* __restrict__ W, _Float16* __restrict__ WT,
                        int Kd, int Nd, int Kp, int nmat) {
  size_t idx = (size_t)blockIdx.x * blockDim.x + threadIdx.x;
  size_t per = (size_t)Nd * Kp;
  if (idx >= per * (size_t)nmat) return;
  int    e = (int)(idx / per);
  size_t r = idx % per;
  int n = (int)(r / Kp), k = (int)(r % Kp);
  float v = (k < Kd) ? W[(size_t)e * Kd * Nd + (size_t)k * Nd + n] : 0.0f;
  WT[idx] = (_Float16)v;
}

__global__ void k_conv(const float* __restrict__ X, _Float16* __restrict__ Y, size_t total) {
  size_t idx = (size_t)blockIdx.x * blockDim.x + threadIdx.x;
  if (idx < total) Y[idx] = (_Float16)X[idx];
}

// =====================================================================
// WMMA helpers (wave32 fragment layouts per CDNA5 ISA)
// =====================================================================
DEV_INLINE v16h load_frag(const _Float16* p) {
  half8 lo = *(const half8*)p;
  half8 hi = *(const half8*)(p + 16);
  v16h f;
#pragma unroll
  for (int i = 0; i < 8; ++i) { f[i] = lo[i]; f[i + 8] = hi[i]; }
  return f;
}

DEV_INLINE v8f wmma32(v16h a, v16h b, v8f c) {
  return __builtin_amdgcn_wmma_f32_16x16x32_f16(false, a, false, b, (short)0, c, false, false);
}

// acts: 0=none, 1=relu, 2=sigmoid
DEV_INLINE void wmma_epilogue(v8f acc, const float* __restrict__ bias,
                              float* __restrict__ Cf, int ldc, int addResid,
                              _Float16* __restrict__ Ch, int ldch,
                              int row0, int col0, int act, int lane) {
  const int col   = col0 + (lane & 15);
  const float bv  = bias ? bias[col] : 0.0f;
  const int rbase = row0 + ((lane >> 4) << 3);
#pragma unroll
  for (int r = 0; r < 8; ++r) {
    const int row = rbase + r;
    float v = acc[r] + bv;
    if (addResid) v += Cf[(size_t)row * ldc + col];
    if (act == 1)      v = fmaxf(v, 0.0f);
    else if (act == 2) v = 1.0f / (1.0f + __expf(-v));
    if (Cf) Cf[(size_t)row * ldc + col]  = v;
    if (Ch) Ch[(size_t)row * ldch + col] = (_Float16)v;
  }
}

// 2x4 register-blocked core: one wave owns a 32x64 tile (8 independent
// accumulators keep the XDL pipe fed with hazard-free back-to-back WMMAs).
DEV_INLINE void wmma_block24(const _Float16* __restrict__ A, int lda,
                             const _Float16* __restrict__ BT, int ldb,
                             int row0, int col0, int Kp, int lane, v8f acc[2][4]) {
  const int koff = (lane & 16) ? 8 : 0;
  const _Float16* pA0 = A + (size_t)(row0 + (lane & 15)) * lda + koff;
  const _Float16* pA1 = pA0 + (size_t)16 * lda;
  const _Float16* pB[4];
#pragma unroll
  for (int n = 0; n < 4; ++n)
    pB[n] = BT + (size_t)(col0 + n * 16 + (lane & 15)) * ldb + koff;
  for (int k0 = 0; k0 < Kp; k0 += 32) {
    __builtin_prefetch(pA0 + k0 + 128, 0, 1);
    __builtin_prefetch(pA1 + k0 + 128, 0, 1);
    v16h a0 = load_frag(pA0 + k0);
    v16h a1 = load_frag(pA1 + k0);
    v16h bf[4];
#pragma unroll
    for (int n = 0; n < 4; ++n) bf[n] = load_frag(pB[n] + k0);
#pragma unroll
    for (int n = 0; n < 4; ++n) {
      acc[0][n] = wmma32(a0, bf[n], acc[0][n]);
      acc[1][n] = wmma32(a1, bf[n], acc[1][n]);
    }
  }
}

// =====================================================================
// Tensor Data Mover: stage a 64 x KC f16 tile (strided rows in global)
// into LDS with one DMA instruction. D# packing per CDNA5 ISA.
// =====================================================================
#if HAVE_TDM
DEV_INLINE void tdm_load_2d(unsigned lds_addr, const void* gptr,
                            unsigned rows, unsigned cols, unsigned row_stride) {
  unsigned long long ga = (unsigned long long)(size_t)gptr;
  u32x4 g0; i32x8 g1; i32x4 g2 = {}; i32x4 g3 = {};
  g0[0] = 1u;                                          // count=1 (valid user D#)
  g0[1] = lds_addr;                                    // lds_addr (bytes)
  g0[2] = (unsigned)(ga & 0xffffffffu);                // global_addr[31:0]
  g0[3] = (unsigned)((ga >> 32) & 0x1ffffffu) | (2u << 30); // addr[56:32] | type=2
  g1[0] = (int)(1u << 16);                             // data_size=1 (2 bytes)
  g1[1] = (int)((cols & 0xffffu) << 16);               // tensor_dim0[15:0]
  g1[2] = (int)(((cols >> 16) & 0xffffu) | ((rows & 0xffffu) << 16)); // dim0 hi | dim1 lo
  g1[3] = (int)(((rows >> 16) & 0xffffu) | ((cols & 0xffffu) << 16)); // dim1 hi | tile_dim0
  g1[4] = (int)(rows & 0xffffu);                       // tile_dim1 (tile_dim2=0)
  g1[5] = (int)row_stride;                             // tensor_dim0_stride[31:0]
  g1[6] = 0; g1[7] = 0;
#if __clang_major__ >= 23
  i32x8 g4 = {};
  __builtin_amdgcn_tensor_load_to_lds(g0, g1, g2, g3, g4, 0);
#else
  __builtin_amdgcn_tensor_load_to_lds(g0, g1, g2, g3, 0);
#endif
}
#endif

// =====================================================================
// Large-GEMM kernel: 256x64 block tile, 8 waves x (32x64). B panel staged
// through LDS by the TDM (double buffered, s_wait_tensorcnt), A from global.
// =====================================================================
__global__ __launch_bounds__(256)
void k_tdm_gemm(const _Float16* __restrict__ A, int lda,
                const _Float16* __restrict__ BT, int ldb,
                const float* __restrict__ bias,
                float* __restrict__ Cf, int ldc, int addResid,
                _Float16* __restrict__ Ch, int ldch,
                int Kp, int act) {
  __shared__ _Float16 sB[2][64 * KC];
  const int lane = threadIdx.x & 31;
  const int wave = threadIdx.x >> 5;
  const int col0 = blockIdx.x * 64;
  const int row0 = blockIdx.y * 256 + wave * 32;
  const int koff = (lane & 16) ? 8 : 0;
  const _Float16* pA0 = A + (size_t)(row0 + (lane & 15)) * lda + koff;
  const _Float16* pA1 = pA0 + (size_t)16 * lda;
  v8f acc[2][4] = {};
  const int nChunks = Kp / KC;
  const _Float16* Bbase = BT + (size_t)col0 * ldb;
#if HAVE_TDM
  if (threadIdx.x == 0)
    tdm_load_2d((unsigned)(size_t)&sB[0][0], Bbase, 64, KC, (unsigned)ldb);
#endif
  for (int c = 0; c < nChunks; ++c) {
    const int cur = c & 1;
#if HAVE_TDM
    if (threadIdx.x == 0) {
      if (c + 1 < nChunks) {
        tdm_load_2d((unsigned)(size_t)&sB[cur ^ 1][0], Bbase + (c + 1) * KC,
                    64, KC, (unsigned)ldb);
        __builtin_amdgcn_s_wait_tensorcnt(1);   // chunk c complete, c+1 in flight
      } else {
        __builtin_amdgcn_s_wait_tensorcnt(0);
      }
    }
    __syncthreads();
#else
    for (int idx = threadIdx.x; idx < 64 * KC; idx += 256) {
      int r = idx / KC, k = idx % KC;
      sB[cur][idx] = BT[(size_t)(col0 + r) * ldb + c * KC + k];
    }
    __syncthreads();
#endif
#pragma unroll
    for (int kk = 0; kk < KC; kk += 32) {
      const int k0 = c * KC + kk;
      __builtin_prefetch(pA0 + k0 + 128, 0, 1);
      __builtin_prefetch(pA1 + k0 + 128, 0, 1);
      v16h a0 = load_frag(pA0 + k0);
      v16h a1 = load_frag(pA1 + k0);
      v16h bf[4];
#pragma unroll
      for (int n = 0; n < 4; ++n)
        bf[n] = load_frag(&sB[cur][(n * 16 + (lane & 15)) * KC + kk + koff]);
#pragma unroll
      for (int n = 0; n < 4; ++n) {
        acc[0][n] = wmma32(a0, bf[n], acc[0][n]);
        acc[1][n] = wmma32(a1, bf[n], acc[1][n]);
      }
    }
    __syncthreads();
  }
#pragma unroll
  for (int m = 0; m < 2; ++m)
#pragma unroll
    for (int n = 0; n < 4; ++n)
      wmma_epilogue(acc[m][n], bias, Cf, ldc, addResid, Ch, ldch,
                    row0 + m * 16, col0 + n * 16, act, lane);
}

// mid-size: one wave = 32x64 tile, operands straight from global/L2
__global__ __launch_bounds__(32)
void k_gemm_b(const _Float16* __restrict__ A, int lda,
              const _Float16* __restrict__ BT, int ldb,
              const float* __restrict__ bias,
              float* __restrict__ Cf, int ldc, int addResid,
              _Float16* __restrict__ Ch, int ldch, int Kp, int act) {
  const int lane = threadIdx.x;
  const int row0 = blockIdx.y * 32, col0 = blockIdx.x * 64;
  v8f acc[2][4] = {};
  wmma_block24(A, lda, BT, ldb, row0, col0, Kp, lane, acc);
#pragma unroll
  for (int m = 0; m < 2; ++m)
#pragma unroll
    for (int n = 0; n < 4; ++n)
      wmma_epilogue(acc[m][n], bias, Cf, ldc, addResid, Ch, ldch,
                    row0 + m * 16, col0 + n * 16, act, lane);
}

// small-N fallback: one wave = 16x16 tile
__global__ __launch_bounds__(32)
void k_gemm(const _Float16* __restrict__ A, int lda,
            const _Float16* __restrict__ BT, int ldb,
            const float* __restrict__ bias,
            float* __restrict__ Cf, int ldc, int addResid,
            _Float16* __restrict__ Ch, int ldch, int Kp, int act) {
  const int lane = threadIdx.x;
  const int row0 = blockIdx.y * 16, col0 = blockIdx.x * 16;
  const int koff = (lane & 16) ? 8 : 0;
  const _Float16* pA = A + (size_t)(row0 + (lane & 15)) * lda + koff;
  const _Float16* pB = BT + (size_t)(col0 + (lane & 15)) * ldb + koff;
  v8f acc = {};
  for (int k0 = 0; k0 < Kp; k0 += 32)
    acc = wmma32(load_frag(pA + k0), load_frag(pB + k0), acc);
  wmma_epilogue(acc, bias, Cf, ldc, addResid, Ch, ldch, row0, col0, act, lane);
}

// Expert-gathered batched GEMM, 32x64 per wave (sample = blockIdx.z)
__global__ __launch_bounds__(32)
void k_gemm_expert_b(const _Float16* __restrict__ A, int lda,
                     const _Float16* __restrict__ Wpool, int ldb, size_t wstride,
                     const float* __restrict__ biasPool, int bstride,
                     const int* __restrict__ ids,
                     float* __restrict__ Cf, int ldc,
                     _Float16* __restrict__ Ch, int ldch, int Kp, int act) {
  const int lane = threadIdx.x;
  const int e    = ids[blockIdx.z];
  const _Float16* BT = Wpool + (size_t)e * wstride;
  const float* bias  = biasPool + (size_t)e * bstride;
  const int row0 = blockIdx.z * Tn;
  const int col0 = blockIdx.x * 64;
  v8f acc[2][4] = {};
  wmma_block24(A, lda, BT, ldb, row0, col0, Kp, lane, acc);
#pragma unroll
  for (int m = 0; m < 2; ++m)
#pragma unroll
    for (int n = 0; n < 4; ++n)
      wmma_epilogue(acc[m][n], bias, Cf, ldc, 0, Ch, ldch,
                    row0 + m * 16, col0 + n * 16, act, lane);
}

// Expert small-N (N=16): one wave = 16x16
__global__ __launch_bounds__(32)
void k_gemm_expert(const _Float16* __restrict__ A, int lda,
                   const _Float16* __restrict__ Wpool, int ldb, size_t wstride,
                   const float* __restrict__ biasPool, int bstride,
                   const int* __restrict__ ids,
                   float* __restrict__ Cf, int ldc,
                   _Float16* __restrict__ Ch, int ldch, int Kp, int act) {
  const int lane = threadIdx.x;
  const int e    = ids[blockIdx.z];
  const _Float16* BT = Wpool + (size_t)e * wstride;
  const float* bias  = biasPool + (size_t)e * bstride;
  const int row0 = blockIdx.z * Tn + blockIdx.y * 16;
  const int col0 = blockIdx.x * 16;
  const int koff = (lane & 16) ? 8 : 0;
  const _Float16* pA = A + (size_t)(row0 + (lane & 15)) * lda + koff;
  const _Float16* pB = BT + (size_t)(col0 + (lane & 15)) * ldb + koff;
  v8f acc = {};
  for (int k0 = 0; k0 < Kp; k0 += 32)
    acc = wmma32(load_frag(pA + k0), load_frag(pB + k0), acc);
  wmma_epilogue(acc, bias, Cf, ldc, 0, Ch, ldch, row0, col0, act, lane);
}

// =====================================================================
// temporal embedding: te[T,128] = relu(ts@w1+b1)@w2+b2, ts = t/31
// =====================================================================
__global__ void k_temporal(const float* __restrict__ w1, const float* __restrict__ b1,
                           const float* __restrict__ w2, const float* __restrict__ b2,
                           float* __restrict__ te) {
  __shared__ float h[128];
  const int t = blockIdx.x, j = threadIdx.x;
  const float ts = (float)t / 31.0f;
  h[j] = fmaxf(ts * w1[j] + b1[j], 0.0f);
  __syncthreads();
  float acc = b2[j];
  for (int i = 0; i < 128; ++i) acc += h[i] * w2[i * 128 + j];
  te[t * 128 + j] = acc;
}

__global__ void k_build_cat(const _Float16* __restrict__ se16, const float* __restrict__ emb,
                            const int* __restrict__ ids, _Float16* __restrict__ cat) {
  size_t idx = (size_t)blockIdx.x * blockDim.x + threadIdx.x;
  if (idx >= (size_t)Bx * 576) return;
  int b = (int)(idx / 576), c = (int)(idx % 576);
  cat[idx] = (c < 512) ? se16[(size_t)b * 512 + c]
                       : (_Float16)emb[(size_t)ids[b] * 64 + (c - 512)];
}

__global__ void k_build_x(const float* __restrict__ se, const float* __restrict__ emb,
                          const int* __restrict__ ids, const float* __restrict__ te,
                          float* __restrict__ X, _Float16* __restrict__ X16) {
  size_t idx = (size_t)blockIdx.x * blockDim.x + threadIdx.x;
  if (idx >= (size_t)TOK * Dm) return;
  int c = (int)(idx % Dm);
  size_t rt = idx / Dm;
  int b = (int)(rt / Tn), t = (int)(rt % Tn);
  float v;
  if (c < 512)      v = se[(size_t)b * 512 + c];
  else if (c < 576) v = emb[(size_t)ids[b] * 64 + (c - 512)];
  else              v = te[t * 128 + (c - 576)];
  X[idx] = v; X16[idx] = (_Float16)v;
}

// attention for one (b,h): T=32, HD=88. One wave, lane = query index.
__global__ __launch_bounds__(32)
void k_attention(const _Float16* __restrict__ Qa, const _Float16* __restrict__ Ka,
                 const _Float16* __restrict__ Va, _Float16* __restrict__ O) {
  __shared__ float sK[Tn * HDn];
  __shared__ float sV[Tn * HDn];
  const int b = blockIdx.y, h = blockIdx.x, lane = threadIdx.x;
  const size_t base = (size_t)b * Tn * Dm + (size_t)h * HDn;
  for (int i = lane; i < Tn * HDn; i += 32) {
    int kk = i / HDn, d = i % HDn;
    sK[i] = (float)Ka[base + (size_t)kk * Dm + d];
    sV[i] = (float)Va[base + (size_t)kk * Dm + d];
  }
  __syncthreads();
  float q[HDn];
#pragma unroll
  for (int d = 0; d < HDn; ++d) q[d] = (float)Qa[base + (size_t)lane * Dm + d];
  float s[Tn]; float mx = -1e30f;
  for (int k = 0; k < Tn; ++k) {
    float a = 0.0f;
    for (int d = 0; d < HDn; ++d) a += q[d] * sK[k * HDn + d];
    a *= 0.1066003581778052f; // 1/sqrt(88)
    s[k] = a; mx = fmaxf(mx, a);
  }
  float den = 0.0f;
  for (int k = 0; k < Tn; ++k) { s[k] = __expf(s[k] - mx); den += s[k]; }
  const float inv = 1.0f / den;
  for (int d = 0; d < HDn; ++d) {
    float o = 0.0f;
    for (int k = 0; k < Tn; ++k) o += s[k] * sV[k * HDn + d];
    O[base + (size_t)lane * Dm + d] = (_Float16)(o * inv);
  }
}

// layernorm over D=704 rows of X (in place), also refresh f16 copy
__global__ void k_layernorm(float* __restrict__ X, _Float16* __restrict__ X16,
                            const float* __restrict__ g, const float* __restrict__ bta) {
  __shared__ float red[256];
  const int row = blockIdx.x, tid = threadIdx.x;
  float* x = X + (size_t)row * Dm;
  float sum = 0.0f;
  for (int c = tid; c < Dm; c += 256) sum += x[c];
  red[tid] = sum; __syncthreads();
  for (int st = 128; st > 0; st >>= 1) { if (tid < st) red[tid] += red[tid + st]; __syncthreads(); }
  const float mean = red[0] / (float)Dm; __syncthreads();
  float var = 0.0f;
  for (int c = tid; c < Dm; c += 256) { float d = x[c] - mean; var += d * d; }
  red[tid] = var; __syncthreads();
  for (int st = 128; st > 0; st >>= 1) { if (tid < st) red[tid] += red[tid + st]; __syncthreads(); }
  const float inv = rsqrtf(red[0] / (float)Dm + 1e-5f);
  for (int c = tid; c < Dm; c += 256) {
    float o = (x[c] - mean) * inv * g[c] + bta[c];
    x[c] = o; X16[(size_t)row * Dm + c] = (_Float16)o;
  }
}

// =====================================================================
extern "C" void kernel_launch(void* const* d_in, const int* in_sizes, int n_in,
                              void* d_out, int out_size, void* d_ws, size_t ws_size,
                              hipStream_t stream) {
  (void)in_sizes; (void)n_in; (void)out_size; (void)ws_size;
  auto F = [&](int i) { return (const float*)d_in[i]; };
  const float* ss  = F(0);
  const int*   ids = (const int*)d_in[1];
  const float* ah  = F(2);
  const float *se_w1=F(3),*se_b1=F(4),*se_w2=F(5),*se_b2=F(6);
  const float *cl_w1=F(7),*cl_b1=F(8),*cl_w2=F(9),*cl_b2=F(10),*cl_w3=F(11),*cl_b3=F(12);
  const float *embp=F(13);
  const float *tm_w1=F(14),*tm_b1=F(15),*tm_w2=F(16),*tm_b2=F(17);
  const float *ji_w1=F(18),*ji_b1=F(19),*ji_w2=F(20),*ji_b2=F(21);
  const float *kp_w1=F(22),*kp_b1=F(23),*kp_w2=F(24),*kp_b2=F(25),*kp_w3=F(26),*kp_b3=F(27);
  const float *fx_w1=F(28),*fx_b1=F(29),*fx_w2=F(30),*fx_b2=F(31);
  const float *hi_w1=F(32),*hi_b1=F(33),*hi_w2=F(34),*hi_b2=F(35);
  const float *Lw[3][12];
  for (int l = 0; l < 3; ++l)
    for (int j = 0; j < 12; ++j) Lw[l][j] = F(36 + 12 * l + j);
  // 0 wqkv,1 bqkv,2 wo,3 bo,4 g1,5 b1,6 wf1,7 bf1,8 wf2,9 bf2,10 g2,11 b2

  // ---- workspace carve ----
  char* wsb = (char*)d_ws; size_t off = 0;
  auto alc = [&](size_t elems, size_t esz) -> void* {
    void* p = wsb + off; off += (elems * esz + 255) & ~(size_t)255; return p;
  };
  auto a16 = [&](size_t e) { return (_Float16*)alc(e, 2); };
  auto a32 = [&](size_t e) { return (float*)alc(e, 4); };

  _Float16 *se_w1T=a16(512*64), *se_w2T=a16(512*512);
  _Float16 *cl_w1T=a16(128*512), *cl_w2T=a16(64*128), *cl_w3T=a16(16*64);
  _Float16 *hi_w1T=a16(512*4096), *hi_w2T=a16(512*512);
  _Float16 *ji_w1T=a16(128*576), *ji_w2T=a16(32*128);
  _Float16 *wqkvT[3], *woT[3], *wf1T[3], *wf2T[3];
  for (int l = 0; l < 3; ++l) {
    wqkvT[l]=a16((size_t)2112*704); woT[l]=a16((size_t)704*704);
    wf1T[l]=a16((size_t)512*704);   wf2T[l]=a16((size_t)704*512);
  }
  _Float16 *kpw1T=a16((size_t)En*256*704), *kpw2T=a16((size_t)En*128*256), *kpw3T=a16((size_t)En*16*128);
  _Float16 *fxw1T=a16(128*32), *fxw2T=a16(32*128);
  _Float16 *ss16=a16((size_t)Bx*64), *ah16=a16((size_t)Bx*4096);
  _Float16 *seh16=a16((size_t)Bx*512), *se16=a16((size_t)Bx*512);
  float    *se_f32=a32((size_t)Bx*512);
  _Float16 *clh1=a16((size_t)Bx*128), *clh2=a16((size_t)Bx*64);
  _Float16 *hih16=a16((size_t)Bx*512);
  _Float16 *cat16=a16((size_t)Bx*576), *jih16=a16((size_t)Bx*128);
  float    *te32=a32((size_t)Tn*128);
  float    *XF=a32((size_t)TOK*Dm);
  _Float16 *X16=a16((size_t)TOK*Dm);
  _Float16 *Q16=a16((size_t)TOK*Dm), *Kk16=a16((size_t)TOK*Dm), *V16=a16((size_t)TOK*Dm);
  _Float16 *AT16=a16((size_t)TOK*Dm);
  _Float16 *H16 = AT16;          // FFN hidden reuses AT16 (dead after o-proj)
  _Float16 *E1 = Q16;            // expert phase reuses q/k/v buffers
  _Float16 *E2 = Kk16;
  _Float16 *KJ16 = V16;          // [TOK,32] zero-padded
  _Float16 *FH16 = V16 + (size_t)TOK * 64;

  float* out_full   = (float*)d_out;                       // [B,T,ACT]
  float* out_logits = out_full + (size_t)TOK * ACTn;       // [B,E]
  float* out_ji     = out_logits + (size_t)Bx * En;        // [B,ACT]
  float* out_kj     = out_ji + (size_t)Bx * ACTn;          // [B,T,K]

  auto convT = [&](const float* W, _Float16* WT, int Kd, int Nd, int Kp, int nm) {
    size_t tot = (size_t)Nd * Kp * nm;
    k_convT<<<(unsigned)((tot + 255) / 256), 256, 0, stream>>>(W, WT, Kd, Nd, Kp, nm);
  };
  auto conv = [&](const float* X, _Float16* Y, size_t tot) {
    k_conv<<<(unsigned)((tot + 255) / 256), 256, 0, stream>>>(X, Y, tot);
  };
  // route by shape: TDM-staged block kernel for big GEMMs, wave-blocked or
  // 16x16 fallback otherwise
  auto gemm = [&](const _Float16* A, int lda, const _Float16* BT, int ldb, const float* bias,
                  float* Cf, int ldc, int addR, _Float16* Ch, int ldch,
                  int M, int N, int Kp, int act) {
    if (M % 256 == 0 && N % 64 == 0 && Kp % KC == 0)
      k_tdm_gemm<<<dim3(N / 64, M / 256), 256, 0, stream>>>(A, lda, BT, ldb, bias, Cf, ldc, addR, Ch, ldch, Kp, act);
    else if (M % 32 == 0 && N % 64 == 0)
      k_gemm_b<<<dim3(N / 64, M / 32), 32, 0, stream>>>(A, lda, BT, ldb, bias, Cf, ldc, addR, Ch, ldch, Kp, act);
    else
      k_gemm<<<dim3(N / 16, M / 16), 32, 0, stream>>>(A, lda, BT, ldb, bias, Cf, ldc, addR, Ch, ldch, Kp, act);
  };

  // ---- stage weights to f16 (transposed, K padded to 32) ----
  convT(se_w1, se_w1T, 64, 512, 64, 1);     convT(se_w2, se_w2T, 512, 512, 512, 1);
  convT(cl_w1, cl_w1T, 512, 128, 512, 1);   convT(cl_w2, cl_w2T, 128, 64, 128, 1);
  convT(cl_w3, cl_w3T, 64, 16, 64, 1);
  convT(hi_w1, hi_w1T, 4096, 512, 4096, 1); convT(hi_w2, hi_w2T, 512, 512, 512, 1);
  convT(ji_w1, ji_w1T, 576, 128, 576, 1);   convT(ji_w2, ji_w2T, 128, 32, 128, 1);
  for (int l = 0; l < 3; ++l) {
    convT(Lw[l][0], wqkvT[l], 704, 2112, 704, 1);
    convT(Lw[l][2], woT[l],   704, 704,  704, 1);
    convT(Lw[l][6], wf1T[l],  704, 512,  704, 1);
    convT(Lw[l][8], wf2T[l],  512, 704,  512, 1);
  }
  convT(kp_w1, kpw1T, 704, 256, 704, En);
  convT(kp_w2, kpw2T, 256, 128, 256, En);
  convT(kp_w3, kpw3T, 128, 16, 128, En);
  convT(fx_w1, fxw1T, 16, 128, 32, 1);      // K=16 zero padded to 32
  convT(fx_w2, fxw2T, 128, 32, 128, 1);
  conv(ss, ss16, (size_t)Bx * 64);
  conv(ah, ah16, (size_t)Bx * 4096);
  k_temporal<<<Tn, 128, 0, stream>>>(tm_w1, tm_b1, tm_w2, tm_b2, te32);

  // ---- state encoder ----
  gemm(ss16, 64, se_w1T, 64, se_b1, nullptr, 0, 0, seh16, 512, Bx, 512, 64, 1);
  gemm(seh16, 512, se_w2T, 512, se_b2, se_f32, 512, 0, se16, 512, Bx, 512, 512, 0);
  // ---- classifier (pre-history se) ----
  gemm(se16, 512, cl_w1T, 512, cl_b1, nullptr, 0, 0, clh1, 128, Bx, 128, 512, 1);
  gemm(clh1, 128, cl_w2T, 128, cl_b2, nullptr, 0, 0, clh2, 64, Bx, 64, 128, 1);
  gemm(clh2, 64, cl_w3T, 64, cl_b3, out_logits, 16, 0, nullptr, 0, Bx, 16, 64, 0);
  // ---- history encoder, residual into se ----
  gemm(ah16, 4096, hi_w1T, 4096, hi_b1, nullptr, 0, 0, hih16, 512, Bx, 512, 4096, 1);
  gemm(hih16, 512, hi_w2T, 512, hi_b2, se_f32, 512, 1, se16, 512, Bx, 512, 512, 0);
  // ---- joint importance head ----
  k_build_cat<<<(unsigned)(((size_t)Bx * 576 + 255) / 256), 256, 0, stream>>>(se16, embp, ids, cat16);
  gemm(cat16, 576, ji_w1T, 576, ji_b1, nullptr, 0, 0, jih16, 128, Bx, 128, 576, 1);
  gemm(jih16, 128, ji_w2T, 128, ji_b2, out_ji, 32, 0, nullptr, 0, Bx, 32, 128, 2);
  // ---- build temporal sequence ----
  k_build_x<<<(unsigned)(((size_t)TOK * Dm + 255) / 256), 256, 0, stream>>>(se_f32, embp, ids, te32, XF, X16);

  // ---- 3 transformer encoder layers ----
  for (int l = 0; l < 3; ++l) {
    const _Float16* wq = wqkvT[l];
    gemm(X16, Dm, wq,                     Dm, Lw[l][1],        nullptr, 0, 0, Q16,  Dm, TOK, Dm, Dm, 0);
    gemm(X16, Dm, wq + (size_t)704 * Dm,  Dm, Lw[l][1] + 704,  nullptr, 0, 0, Kk16, Dm, TOK, Dm, Dm, 0);
    gemm(X16, Dm, wq + (size_t)1408 * Dm, Dm, Lw[l][1] + 1408, nullptr, 0, 0, V16,  Dm, TOK, Dm, Dm, 0);
    k_attention<<<dim3(NHn, Bx), 32, 0, stream>>>(Q16, Kk16, V16, AT16);
    gemm(AT16, Dm, woT[l], Dm, Lw[l][3], XF, Dm, 1, nullptr, 0, TOK, Dm, Dm, 0);
    k_layernorm<<<TOK, 256, 0, stream>>>(XF, X16, Lw[l][4], Lw[l][5]);
    gemm(X16, Dm, wf1T[l], Dm, Lw[l][7], nullptr, 0, 0, H16, 512, TOK, 512, Dm, 1);
    gemm(H16, 512, wf2T[l], 512, Lw[l][9], XF, Dm, 1, nullptr, 0, TOK, Dm, 512, 0);
    k_layernorm<<<TOK, 256, 0, stream>>>(XF, X16, Lw[l][10], Lw[l][11]);
  }

  // ---- instruction-indexed expert MLP ----
  k_gemm_expert_b<<<dim3(256 / 64, 1, Bx), 32, 0, stream>>>(
      X16, Dm, kpw1T, 704, (size_t)256 * 704, kp_b1, 256, ids, nullptr, 0, E1, 256, 704, 1);
  k_gemm_expert_b<<<dim3(128 / 64, 1, Bx), 32, 0, stream>>>(
      E1, 256, kpw2T, 256, (size_t)128 * 256, kp_b2, 128, ids, nullptr, 0, E2, 128, 256, 1);
  hipMemsetAsync(KJ16, 0, (size_t)TOK * 32 * sizeof(_Float16), stream);
  k_gemm_expert<<<dim3(1, Tn / 16, Bx), 32, 0, stream>>>(
      E2, 128, kpw3T, 128, (size_t)16 * 128, kp_b3, 16, ids, out_kj, 16, KJ16, 32, 128, 0);

  // ---- temporal expander head ----
  gemm(KJ16, 32, fxw1T, 32, fx_b1, nullptr, 0, 0, FH16, 128, TOK, 128, 32, 1);
  gemm(FH16, 128, fxw2T, 128, fx_b2, out_full, 32, 0, nullptr, 0, TOK, 32, 128, 0);
}
